// MultiHeadSelfAttention_57518202028111
// MI455X (gfx1250) — matrix operand order repaired and checked by reading, not compiled, not run on hardware
//
#include <hip/hip_runtime.h>

// ---------------------------------------------------------------------------
// MHA forward for MI455X (gfx1250, wave32, WMMA + TDM async tensor loads).
// Precision: bf16 operands, f32 accumulate (V_WMMA_F32_16X16X32_BF16).
// ---------------------------------------------------------------------------

#define BATCH 2
#define SEQ   2048
#define DIM   1024
#define HEADS 16
#define DHEAD 64
#define BS    (BATCH * SEQ)          // 4096 rows
#define LOG1E4 9.210340371976184f    // ln(10000)

typedef __attribute__((ext_vector_type(16))) __bf16 v16bf;
typedef __attribute__((ext_vector_type(8)))  __bf16 v8bf;
typedef __attribute__((ext_vector_type(8)))  float  v8f;
typedef __attribute__((ext_vector_type(4)))  unsigned int u32x4;
typedef __attribute__((ext_vector_type(4)))  int i32x4;
typedef __attribute__((ext_vector_type(8)))  int i32x8;

// ---------------------------------------------------------------------------
// WMMA helpers (layouts per CDNA5 ISA §7.12.2, wave32)
// ---------------------------------------------------------------------------

__device__ inline v8f wmma_bf16(v16bf a, v16bf b, v8f c) {
  return __builtin_amdgcn_wmma_f32_16x16x32_bf16(false, a, false, b,
                                                 (short)0, c, false, false);
}

// A-matrix 16x32 bf16. base -> element (m=0,k=0), row stride ld (elements).
__device__ inline v16bf load_a_frag(const __bf16* base, int ld) {
  const int lane = threadIdx.x & 31;
  const int m    = lane & 15;
  const int half = lane >> 4;
  const __bf16* p = base + m * ld + half * 16;
  v8bf lo = *(const v8bf*)(p);        // K = half*16 .. +7
  v8bf hi = *(const v8bf*)(p + 8);    // K = half*16+8 .. +15
  return __builtin_shufflevector(lo, hi, 0,1,2,3,4,5,6,7,8,9,10,11,12,13,14,15);
}

// B-matrix 32x16 bf16, B[k][n] = rowsrc[n][k] (rowsrc row-major, stride ld).
__device__ inline v16bf load_b_frag_rows(const __bf16* base, int ld) {
  const int lane  = threadIdx.x & 31;
  const int n     = lane & 15;
  const int khalf = lane >> 4;
  return *(const v16bf*)(base + n * ld + khalf * 16);
}

// ---------------------------------------------------------------------------
// Tensor Data Mover: async 2D bf16 tile load Global -> LDS.
// D# packing per CDNA5 ISA (groups 2/3 = unused).
// All args must be wave-uniform (descriptor lives in SGPRs).
// ---------------------------------------------------------------------------
__device__ inline void tdm_load_2d_bf16(unsigned lds_byte_addr, const void* gptr,
                                        unsigned tensor_d0, unsigned tensor_d1,
                                        unsigned tile_d0, unsigned tile_d1,
                                        unsigned long long stride0 /*elems*/) {
  unsigned long long ga = (unsigned long long)(uintptr_t)gptr;
  u32x4 g0;
  g0[0] = 1u;                                         // count=1 (valid D#)
  g0[1] = lds_byte_addr;                              // lds_addr
  g0[2] = (unsigned)(ga & 0xffffffffu);               // global_addr[31:0]
  g0[3] = (unsigned)((ga >> 32) & 0x01ffffffu)        // global_addr[56:32]
        | 0x80000000u;                                // type=2 ("image")
  i32x8 g1;
  g1[0] = (int)(1u << 16);                            // data_size=1 -> 2 bytes
  g1[1] = (int)((tensor_d0 & 0xffffu) << 16);         // tensor_dim0[15:0]
  g1[2] = (int)(((tensor_d0 >> 16) & 0xffffu)         // tensor_dim0[31:16]
        | ((tensor_d1 & 0xffffu) << 16));             // tensor_dim1[15:0]
  g1[3] = (int)(((tensor_d1 >> 16) & 0xffffu)         // tensor_dim1[31:16]
        | ((tile_d0 & 0xffffu) << 16));               // tile_dim0
  g1[4] = (int)(tile_d1 & 0xffffu);                   // tile_dim1 (tile_dim2=0)
  g1[5] = (int)(unsigned)(stride0 & 0xffffffffu);     // tensor_dim0_stride lo
  g1[6] = (int)(unsigned)((stride0 >> 32) & 0xffffu); // stride hi (dim1_stride=0)
  g1[7] = 0;
  i32x4 z4 = {0, 0, 0, 0};
#if __clang_major__ >= 23
  i32x8 z8 = {0, 0, 0, 0, 0, 0, 0, 0};
  __builtin_amdgcn_tensor_load_to_lds(g0, g1, z4, z4, z8, 0);
#else
  __builtin_amdgcn_tensor_load_to_lds(g0, g1, z4, z4, 0);
#endif
}

// ---------------------------------------------------------------------------
// fp32 -> bf16 converter
// ---------------------------------------------------------------------------
__global__ void cvt_bf16_kernel(const float* __restrict__ in,
                                __bf16* __restrict__ out, int n) {
  int i = blockIdx.x * blockDim.x + threadIdx.x;
  int stride = gridDim.x * blockDim.x;
  for (; i < n; i += stride) out[i] = (__bf16)in[i];
}

// ---------------------------------------------------------------------------
// Projection GEMM:  Y[row, e] = sum_d X[row, d] * W[e, d]
// One wave per block; each wave computes a 32-row x 64-col strip
// (2 A-frags x 4 B-frags -> 8 WMMAs per K-step: 24 B loaded per lane-WMMA).
// mode 0/1: Q/K -> RoPE, store bf16 [B,H,S,Dh]
// mode 2:   V   -> store bf16 transposed [B,H,Dh,S]
// mode 3:   out-proj -> store fp32 [B,S,D]
// ---------------------------------------------------------------------------
__global__ void proj_gemm_kernel(const __bf16* __restrict__ X,
                                 const __bf16* __restrict__ W,
                                 __bf16* __restrict__ out_bf,
                                 float* __restrict__ out_f32,
                                 int mode) {
  const int rowtile = blockIdx.x;         // 0..BS/32-1
  const int ctile64 = blockIdx.y;         // 0..DIM/64-1  (== head for QKV)
  const int lane  = threadIdx.x & 31;
  const int nlane = lane & 15;
  const int mhalf = lane >> 4;

  const int row0 = rowtile * 32;
  const int b    = row0 / SEQ;
  const int s0   = row0 % SEQ;

  v8f acc[2][4];
#pragma unroll
  for (int rh = 0; rh < 2; ++rh)
#pragma unroll
    for (int i = 0; i < 4; ++i)
      acc[rh][i] = (v8f){0.f,0.f,0.f,0.f,0.f,0.f,0.f,0.f};

  const __bf16* Xrow = X + (size_t)row0 * DIM;
  for (int k = 0; k < DIM; k += 32) {
    if (k + 64 < DIM)
      __builtin_prefetch(Xrow + (threadIdx.x & 31) * DIM + k + 64, 0, 1);
    v16bf a0 = load_a_frag(Xrow + k, DIM);
    v16bf a1 = load_a_frag(Xrow + (size_t)16 * DIM + k, DIM);
#pragma unroll
    for (int ct = 0; ct < 4; ++ct) {
      const __bf16* Wr = W + (size_t)(ctile64 * 64 + ct * 16) * DIM + k;
      v16bf bb = load_b_frag_rows(Wr, DIM);
      acc[0][ct] = wmma_bf16(a0, bb, acc[0][ct]);
      acc[1][ct] = wmma_bf16(a1, bb, acc[1][ct]);
    }
  }

#pragma unroll
  for (int rh = 0; rh < 2; ++rh) {
    if (mode == 0 || mode == 1) {
      const int h = ctile64;
#pragma unroll
      for (int ct = 0; ct < 4; ++ct) {
        const int dd = ct * 16 + nlane;
        const float invf = __expf(-(float)(dd & ~1) * (LOG1E4 / (float)DHEAD));
#pragma unroll
        for (int r = 0; r < 8; ++r) {
          float v = acc[rh][ct][r];
          float partner = __shfl_xor(v, 1, 32);   // even<->odd column pair
          const int s = s0 + rh * 16 + r + 8 * mhalf;
          float sn, cs;
          __sincosf((float)s * invf, &sn, &cs);
          float o = ((dd & 1) == 0) ? (v * cs - partner * sn)
                                    : (partner * sn + v * cs);
          out_bf[((size_t)(b * HEADS + h) * SEQ + s) * DHEAD + dd] = (__bf16)o;
        }
      }
    } else if (mode == 2) {
      const int h = ctile64;
#pragma unroll
      for (int ct = 0; ct < 4; ++ct) {
        const int dd = ct * 16 + nlane;
        v8bf pk;
#pragma unroll
        for (int r = 0; r < 8; ++r) pk[r] = (__bf16)acc[rh][ct][r];
        __bf16* dst = out_bf + ((size_t)(b * HEADS + h) * DHEAD + dd) * SEQ
                             + s0 + rh * 16 + 8 * mhalf;
        *(v8bf*)dst = pk;
      }
    } else {
#pragma unroll
      for (int ct = 0; ct < 4; ++ct) {
        const int e = ctile64 * 64 + ct * 16 + nlane;
#pragma unroll
        for (int r = 0; r < 8; ++r) {
          const int row = row0 + rh * 16 + r + 8 * mhalf;
          out_f32[(size_t)row * DIM + e] = acc[rh][ct][r];
        }
      }
    }
  }
}

// ---------------------------------------------------------------------------
// Flash attention (causal, online softmax), TDM double-buffered K/V staging.
// Block = 128 threads = 4 independent waves; each wave owns 16 query rows,
// its own K/V LDS double buffer, and issues its own tensor_load_to_lds chain
// (TDM ops are in-order per wave; s_wait_tensorcnt<=2 => current pair done).
// ---------------------------------------------------------------------------
__global__ void flash_attn_kernel(const __bf16* __restrict__ Q,
                                  const __bf16* __restrict__ K,
                                  const __bf16* __restrict__ Vt,
                                  __bf16* __restrict__ AO) {
  const int wid   = threadIdx.x >> 5;
  const int lane  = threadIdx.x & 31;
  const int nlane = lane & 15;
  const int mhalf = lane >> 4;
  const int qbase = blockIdx.x * 64 + wid * 16;
  const int h = blockIdx.y;
  const int b = blockIdx.z;

  // Per-wave double buffers: K tile 32x64, V tile (transposed) 64x32.
  __shared__ __align__(32) __bf16 Kbuf[4][2][32 * 64];
  __shared__ __align__(32) __bf16 Vbuf[4][2][64 * 32];
  __shared__ __align__(32) __bf16 Pst[4][16][32];

  const __bf16* Qp = Q + ((size_t)(b * HEADS + h) * SEQ + qbase) * DHEAD;
  const v16bf qa0 = load_a_frag(Qp, DHEAD);        // d = 0..31
  const v16bf qa1 = load_a_frag(Qp + 32, DHEAD);   // d = 32..63

  const __bf16* Kbh  = K  + (size_t)(b * HEADS + h) * SEQ * DHEAD;
  const __bf16* Vtbh = Vt + (size_t)(b * HEADS + h) * DHEAD * SEQ;

  // SGPR (wave-uniform) LDS byte offsets for the TDM descriptors.
  unsigned kofs0 = (unsigned)__builtin_amdgcn_readfirstlane(
      (int)(unsigned)(uintptr_t)&Kbuf[wid][0][0]);
  unsigned kofs1 = (unsigned)__builtin_amdgcn_readfirstlane(
      (int)(unsigned)(uintptr_t)&Kbuf[wid][1][0]);
  unsigned vofs0 = (unsigned)__builtin_amdgcn_readfirstlane(
      (int)(unsigned)(uintptr_t)&Vbuf[wid][0][0]);
  unsigned vofs1 = (unsigned)__builtin_amdgcn_readfirstlane(
      (int)(unsigned)(uintptr_t)&Vbuf[wid][1][0]);

  v8f o[4];
#pragma unroll
  for (int i = 0; i < 4; ++i) o[i] = (v8f){0.f,0.f,0.f,0.f,0.f,0.f,0.f,0.f};
  float mrow[8], lrow[8];
#pragma unroll
  for (int r = 0; r < 8; ++r) { mrow[r] = -3.0e38f; lrow[r] = 0.f; }

  const int kvend = qbase + 16;          // keys [0, qbase+15] needed

  // Prologue: async-load first K/V tile pair into buffer 0.
  tdm_load_2d_bf16(kofs0, Kbh, DHEAD, SEQ, DHEAD, 32, DHEAD);
  tdm_load_2d_bf16(vofs0, Vtbh, SEQ, DHEAD, 32, DHEAD, SEQ);

  int buf = 0;
  for (int kv = 0; kv < kvend; kv += 32, buf ^= 1) {
    // Issue next tile pair into the other buffer while we compute.
    const bool more = (kv + 32 < kvend);
    if (more) {
      unsigned ko = buf ? kofs0 : kofs1;
      unsigned vo = buf ? vofs0 : vofs1;
      tdm_load_2d_bf16(ko, Kbh + (size_t)(kv + 32) * DHEAD,
                       DHEAD, SEQ, DHEAD, 32, DHEAD);
      tdm_load_2d_bf16(vo, Vtbh + (kv + 32),
                       SEQ, DHEAD, 32, DHEAD, SEQ);
      __builtin_amdgcn_s_wait_tensorcnt(2);   // current pair complete
    } else {
      __builtin_amdgcn_s_wait_tensorcnt(0);
    }
    asm volatile("" ::: "memory");

    const __bf16* Kl = &Kbuf[wid][buf][0];
    const __bf16* Vl = &Vbuf[wid][buf][0];

    // ---- S = Q*K^T for 32 keys (two 16-key groups) ----
    v8f sf[2];
#pragma unroll
    for (int j = 0; j < 2; ++j) {
      v8f s = (v8f){0.f,0.f,0.f,0.f,0.f,0.f,0.f,0.f};
      s = wmma_bf16(qa0, load_b_frag_rows(Kl + 16 * j * DHEAD,      DHEAD), s);
      s = wmma_bf16(qa1, load_b_frag_rows(Kl + 16 * j * DHEAD + 32, DHEAD), s);
      sf[j] = s;
    }
    // ---- scale + causal mask ----
#pragma unroll
    for (int j = 0; j < 2; ++j) {
      const int kcol = kv + 16 * j + nlane;
#pragma unroll
      for (int r = 0; r < 8; ++r) {
        const int q = qbase + r + 8 * mhalf;
        float v = sf[j][r] * 0.125f;               // 1/sqrt(64)
        sf[j][r] = (kcol > q) ? -3.0e38f : v;
      }
    }
    // ---- row max (butterfly over the 16-lane group) ----
    float rm[8];
#pragma unroll
    for (int r = 0; r < 8; ++r) rm[r] = fmaxf(sf[0][r], sf[1][r]);
#pragma unroll
    for (int off = 1; off < 16; off <<= 1)
#pragma unroll
      for (int r = 0; r < 8; ++r)
        rm[r] = fmaxf(rm[r], __shfl_xor(rm[r], off, 32));
    // ---- online softmax update ----
    float alpha[8], rs[8];
#pragma unroll
    for (int r = 0; r < 8; ++r) {
      float mn = fmaxf(mrow[r], rm[r]);
      alpha[r] = __expf(mrow[r] - mn);
      mrow[r] = mn;
      sf[0][r] = __expf(sf[0][r] - mn);
      sf[1][r] = __expf(sf[1][r] - mn);
      rs[r] = sf[0][r] + sf[1][r];
    }
#pragma unroll
    for (int off = 1; off < 16; off <<= 1)
#pragma unroll
      for (int r = 0; r < 8; ++r)
        rs[r] += __shfl_xor(rs[r], off, 32);
#pragma unroll
    for (int r = 0; r < 8; ++r) lrow[r] = lrow[r] * alpha[r] + rs[r];
#pragma unroll
    for (int i = 0; i < 4; ++i)
#pragma unroll
      for (int r = 0; r < 8; ++r) o[i][r] *= alpha[r];

    // ---- bounce P through LDS: C-layout -> A-layout (16x32 bf16) ----
#pragma unroll
    for (int r = 0; r < 8; ++r) {
      const int m = r + 8 * mhalf;
      Pst[wid][m][nlane]      = (__bf16)sf[0][r];
      Pst[wid][m][16 + nlane] = (__bf16)sf[1][r];
    }
    __builtin_amdgcn_wave_barrier();
    asm volatile("" ::: "memory");
    v16bf pa = load_a_frag(&Pst[wid][0][0], 32);
    __builtin_amdgcn_wave_barrier();
    asm volatile("" ::: "memory");

    // ---- O += P * V  (B[k][n] = Vl[(dtile+n)*32 + k], contiguous 32B) ----
#pragma unroll
    for (int i = 0; i < 4; ++i) {
      v16bf vb = load_b_frag_rows(Vl + i * 16 * 32, 32);
      o[i] = wmma_bf16(pa, vb, o[i]);
    }
  }

  // ---- finalize: O /= l, store bf16 [B,S,D] ----
  float inv[8];
#pragma unroll
  for (int r = 0; r < 8; ++r) inv[r] = 1.0f / lrow[r];
#pragma unroll
  for (int i = 0; i < 4; ++i) {
    const int e = h * DHEAD + i * 16 + nlane;
#pragma unroll
    for (int r = 0; r < 8; ++r) {
      const int s = qbase + r + 8 * mhalf;
      AO[((size_t)b * SEQ + s) * DIM + e] = (__bf16)(o[i][r] * inv[r]);
    }
  }
}

// ---------------------------------------------------------------------------
// Launch
// ---------------------------------------------------------------------------
extern "C" void kernel_launch(void* const* d_in, const int* in_sizes, int n_in,
                              void* d_out, int out_size, void* d_ws, size_t ws_size,
                              hipStream_t stream) {
  const float* X  = (const float*)d_in[0];
  // d_in[1] = token_positions (== arange(S), implicit in kernels)
  const float* Wq = (const float*)d_in[2];
  const float* Wk = (const float*)d_in[3];
  const float* Wv = (const float*)d_in[4];
  const float* Wo = (const float*)d_in[5];
  float* out = (float*)d_out;

  char* ws = (char*)d_ws;
  const size_t XN = (size_t)BS * DIM;      // 4M elems
  const size_t WN = (size_t)DIM * DIM;     // 1M elems
  __bf16* Xbf = (__bf16*)(ws);                         // 8 MiB
  __bf16* Wqb = (__bf16*)(ws + XN * 2);                // 2 MiB each
  __bf16* Wkb = Wqb + WN;
  __bf16* Wvb = Wkb + WN;
  __bf16* Wob = Wvb + WN;
  __bf16* Qr  = Wob + WN;                              // [B,H,S,Dh]
  __bf16* Kr  = Qr + XN;                               // [B,H,S,Dh]
  __bf16* Vt  = Kr + XN;                               // [B,H,Dh,S]
  __bf16* AO  = Vt + XN;                               // [B,S,D]

  // 1) fp32 -> bf16
  cvt_bf16_kernel<<<dim3(4096), dim3(256), 0, stream>>>(X,  Xbf, (int)XN);
  cvt_bf16_kernel<<<dim3(1024), dim3(256), 0, stream>>>(Wq, Wqb, (int)WN);
  cvt_bf16_kernel<<<dim3(1024), dim3(256), 0, stream>>>(Wk, Wkb, (int)WN);
  cvt_bf16_kernel<<<dim3(1024), dim3(256), 0, stream>>>(Wv, Wvb, (int)WN);
  cvt_bf16_kernel<<<dim3(1024), dim3(256), 0, stream>>>(Wo, Wob, (int)WN);

  // 2) Q/K/V projections (RoPE fused; V transposed)
  dim3 pg(BS / 32, DIM / 64);
  proj_gemm_kernel<<<pg, dim3(32), 0, stream>>>(Xbf, Wqb, Qr, nullptr, 0);
  proj_gemm_kernel<<<pg, dim3(32), 0, stream>>>(Xbf, Wkb, Kr, nullptr, 1);
  proj_gemm_kernel<<<pg, dim3(32), 0, stream>>>(Xbf, Wvb, Vt, nullptr, 2);

  // 3) causal flash attention (TDM double-buffered K/V)
  flash_attn_kernel<<<dim3(SEQ / 64, HEADS, BATCH), dim3(128), 0, stream>>>(
      Qr, Kr, Vt, AO);

  // 4) output projection (fp32 store to d_out)
  proj_gemm_kernel<<<pg, dim3(32), 0, stream>>>(AO, Wob, nullptr, out, 3);
}
